// SimpleLSTM_35296041238860
// MI455X (gfx1250) — compile-verified, matrix operand order checked by reading, not run on hardware
//
#include <hip/hip_runtime.h>
#include <hip/hip_bf16.h>
#include <stdint.h>

// Problem constants (from reference)
#define Bn   64
#define Tn   2048
#define Fn   128
#define Un   128
#define G4   512          // 4*UNITS
#define FDn  64
#define BT   (Bn * Tn)    // 131072

typedef __attribute__((ext_vector_type(16))) __bf16 v16bf;
typedef __attribute__((ext_vector_type(8)))  float  v8f;

struct Frag16 { union { v16bf v; uint4 q[2]; unsigned short s[16]; }; };

__device__ __forceinline__ unsigned short f2bf(float x) {
  unsigned u = __float_as_uint(x);
  u += 0x7FFFu + ((u >> 16) & 1u);   // round-to-nearest-even
  return (unsigned short)(u >> 16);
}
__device__ __forceinline__ float sigm(float x) { return 1.0f / (1.0f + __expf(-x)); }
__device__ __forceinline__ float tanh_fast(float x) {
  x = fminf(fmaxf(x, -15.0f), 15.0f);
  float e = __expf(2.0f * x);
  return (e - 1.0f) / (e + 1.0f);
}

// ---------------------------------------------------------------------------
// TDM support: this toolchain exposes the 6-arg clang-23 builtin
//   (uint32x4 g0, int32x8 g1, int32x4 g2, int32x4 g3, int32x8 extra, i32 cpol)
// with fallback to a cooperative copy if the builtin is absent.
// ---------------------------------------------------------------------------
#if __has_builtin(__builtin_amdgcn_tensor_load_to_lds) && \
    __has_builtin(__builtin_amdgcn_s_wait_tensorcnt)
#define HAVE_TDM 1
typedef unsigned int u32x4 __attribute__((ext_vector_type(4)));
typedef int          i32x8 __attribute__((ext_vector_type(8)));
typedef int          i32x4 __attribute__((ext_vector_type(4)));

// 2D TDM tile load: 16 rows x 512 fp32, row stride Tn*G4 elements, into LDS.
// Descriptor packing per CDNA5 ISA.
__device__ __forceinline__ void tdm_load_tile(const float* gptr, unsigned lds_off) {
  const unsigned long long ga = (unsigned long long)(uintptr_t)gptr;
  u32x4 g0;
  g0[0] = 1u;                                        // count=1 (valid user D#)
  g0[1] = lds_off;                                   // lds_addr (bytes)
  g0[2] = (unsigned)(ga & 0xFFFFFFFFu);              // global_addr[31:0]
  g0[3] = (unsigned)((ga >> 32) & 0x01FFFFFFu)       // global_addr[56:32]
        | 0x80000000u;                               // type=2 ("image")
  i32x8 g1;
  g1[0] = (int)(2u << 16);                           // data_size = 4 bytes
  g1[1] = (int)((unsigned)G4 << 16);                 // tensor_dim0 = 512
  g1[2] = (int)(16u << 16);                          // tensor_dim1 = 16
  g1[3] = (int)((unsigned)G4 << 16);                 // tile_dim0 = 512
  g1[4] = (int)16;                                   // tile_dim1 = 16
  g1[5] = (int)(Tn * G4);                            // tensor_dim0_stride lo
  g1[6] = 0;                                         // stride hi / dim1_stride lo
  g1[7] = 0;
  i32x4 gz4 = {0, 0, 0, 0};
  i32x8 gz8 = {0, 0, 0, 0, 0, 0, 0, 0};
  __builtin_amdgcn_tensor_load_to_lds(g0, g1, gz4, gz4, gz8, 0);
}
#else
#define HAVE_TDM 0
#endif

// ---------------------------------------------------------------------------
// Kernel 1: per-row mask = any(x != -1) over F; write masked x as bf16 (ushort)
// and mask as float. One wave (32 lanes) per row; lane covers 4 features.
// ---------------------------------------------------------------------------
__global__ void prep_x_kernel(const float* __restrict__ x,
                              unsigned short* __restrict__ xB,
                              float* __restrict__ maskf) {
  const int row  = blockIdx.x * 8 + (threadIdx.x >> 5);
  const int lane = threadIdx.x & 31;
  const float4 v = *(const float4*)(x + (size_t)row * Fn + lane * 4);
  int flag = (v.x != -1.0f) | (v.y != -1.0f) | (v.z != -1.0f) | (v.w != -1.0f);
  #pragma unroll
  for (int o = 16; o > 0; o >>= 1) flag |= __shfl_xor(flag, o);
  const float m = flag ? 1.0f : 0.0f;
  ushort4 out;
  out.x = f2bf(v.x * m); out.y = f2bf(v.y * m);
  out.z = f2bf(v.z * m); out.w = f2bf(v.w * m);
  *(ushort4*)(xB + (size_t)row * Fn + lane * 4) = out;
  if (lane == 0) maskf[row] = m;
}

// ---------------------------------------------------------------------------
// Kernel 2: transpose+convert W (F x 4U) and U (U x 4U) to n-major bf16
// Wt[n][k], Ut[n][k]  (leading dim 128) so B fragments are contiguous b128s.
// ---------------------------------------------------------------------------
__global__ void prep_wu_kernel(const float* __restrict__ W,
                               const float* __restrict__ Uw,
                               unsigned short* __restrict__ Wt,
                               unsigned short* __restrict__ Ut) {
  const int idx = blockIdx.x * blockDim.x + threadIdx.x;   // 0 .. 131071
  const int sel = idx >> 16;            // 0: W, 1: U
  const int i   = idx & 65535;          // n*128 + k
  const int n   = i >> 7;
  const int k   = i & 127;
  if (sel == 0) Wt[i] = f2bf(W[(size_t)k * G4 + n]);
  else          Ut[i] = f2bf(Uw[(size_t)k * G4 + n]);
}

// ---------------------------------------------------------------------------
// Kernel 3: big GEMM  xW[b*T+t][512] = x_bf16 @ W_bf16 + bias  (fp32 accum).
// One block = one 16-row tile of (B*T); 8 waves, each wave owns 4 N-tiles.
// ---------------------------------------------------------------------------
__global__ void xw_gemm_kernel(const unsigned short* __restrict__ xB,
                               const unsigned short* __restrict__ Wt,
                               const float* __restrict__ bias,
                               float* __restrict__ xW) {
  const int tile = blockIdx.x;            // 0 .. BT/16-1
  const int wave = threadIdx.x >> 5;      // 0..7
  const int lane = threadIdx.x & 31;
  const int nl   = lane & 15;
  const int hi   = (lane >> 4) & 1;

  v8f acc[4];
  #pragma unroll
  for (int j = 0; j < 4; ++j) {
    const float bv = bias[(wave * 4 + j) * 16 + nl];
    #pragma unroll
    for (int r = 0; r < 8; ++r) acc[j][r] = bv;
  }

  const unsigned short* arow = xB + ((size_t)tile * 16 + nl) * Fn;
  #pragma unroll
  for (int kk = 0; kk < 4; ++kk) {
    const int k0 = kk * 32 + hi * 8;
    Frag16 a;
    a.q[0] = *(const uint4*)(arow + k0);
    a.q[1] = *(const uint4*)(arow + k0 + 16);
    #pragma unroll
    for (int j = 0; j < 4; ++j) {
      const int n = (wave * 4 + j) * 16 + nl;
      const unsigned short* bp = Wt + (size_t)n * Fn + k0;
      Frag16 bf;
      bf.q[0] = *(const uint4*)(bp);
      bf.q[1] = *(const uint4*)(bp + 16);
      acc[j] = __builtin_amdgcn_wmma_f32_16x16x32_bf16(
          false, a.v, false, bf.v, (short)0, acc[j], false, false);
    }
  }

  #pragma unroll
  for (int j = 0; j < 4; ++j) {
    const int n = (wave * 4 + j) * 16 + nl;
    #pragma unroll
    for (int r = 0; r < 8; ++r) {
      const size_t row = (size_t)tile * 16 + hi * 8 + r;
      xW[row * G4 + n] = acc[j][r];
    }
  }
}

// ---------------------------------------------------------------------------
// Kernel 4: persistent LSTM recurrence. 4 blocks, each owns 16 batch rows.
// h,c state in LDS. Per step: z = hB @ Ut (WMMA, acc init from TDM-prefetched
// xW tile in LDS), gates + masked state update, h history to global.
// LDS row padding avoids ds bank conflicts (row stride 68/4-dword patterns).
// ---------------------------------------------------------------------------
#define HBP  136    // hB row pad: 272B => lane stride 68 dwords => 4-bank step
#define ZSP  516    // zS row pad

__global__ void lstm_recur_kernel(const float* __restrict__ xW,
                                  const unsigned short* __restrict__ Ut,
                                  const float* __restrict__ maskf,
                                  float* __restrict__ h_hist) {
  __shared__ __align__(16) float          xwL[2][16][G4];   // 64 KB double buffer
  __shared__ __align__(16) float          zS[16][ZSP];      // ~33 KB
  __shared__ __align__(16) float          cS[16][Un];       // 8 KB
  __shared__ __align__(16) float          hS[16][Un];       // 8 KB
  __shared__ __align__(16) unsigned short hB[16][HBP];      // ~4.3 KB

  const int chunk = blockIdx.x;           // 0..3 (16 batch rows each)
  const int tid   = threadIdx.x;
  const int wave  = tid >> 5;
  const int lane  = tid & 31;
  const int nl    = lane & 15;
  const int hi    = (lane >> 4) & 1;

  for (int i = tid; i < 16 * Un; i += 256) {
    ((float*)cS)[i] = 0.0f;
    ((float*)hS)[i] = 0.0f;
  }
  for (int i = tid; i < 16 * HBP; i += 256) ((unsigned short*)hB)[i] = 0;

  // Base pointer of this chunk's rows inside xW: row b = chunk*16 + r,
  // element (b*Tn + t)*G4 + n  ==>  tile origin for step t is xwc + t*G4.
  const float* xwc = xW + (size_t)(chunk * 16) * Tn * G4;

#if HAVE_TDM
  // Prefetch tile t=0 into buffer 0 (wave 0 issues the DMA; EXEC ignored).
  if (wave == 0) {
    tdm_load_tile(xwc, (unsigned)(uintptr_t)&xwL[0][0][0]);
    __builtin_amdgcn_s_wait_tensorcnt(0);
  }
#else
  {
    #pragma unroll
    for (int q = 0; q < 8; ++q) {
      const int fi  = tid * 8 + q;          // 2048 float4 chunks
      const int r   = fi >> 7;
      const int c4  = fi & 127;
      *(float4*)&xwL[0][r][c4 * 4] =
          *(const float4*)(xwc + (size_t)r * Tn * G4 + c4 * 4);
    }
  }
#endif
  __syncthreads();

  for (int t = 0; t < Tn; ++t) {
    const int buf  = t & 1;
    const int nbuf = buf ^ 1;

#if HAVE_TDM
    // Kick off DMA of next step's tile while we compute this one.
    if (wave == 0 && (t + 1) < Tn) {
      tdm_load_tile(xwc + (size_t)(t + 1) * G4,
                    (unsigned)(uintptr_t)&xwL[nbuf][0][0]);
    }
#endif

    // ---- acc init from LDS-resident xW tile (bias already folded in) ----
    v8f acc[4];
    #pragma unroll
    for (int j = 0; j < 4; ++j) {
      const int n = (wave * 4 + j) * 16 + nl;
      #pragma unroll
      for (int r = 0; r < 8; ++r) acc[j][r] = xwL[buf][r + hi * 8][n];
    }

#if !HAVE_TDM
    // Fallback: cooperative double-buffered copy of next tile.
    if ((t + 1) < Tn) {
      #pragma unroll
      for (int q = 0; q < 8; ++q) {
        const int fi = tid * 8 + q;
        const int r  = fi >> 7;
        const int c4 = fi & 127;
        *(float4*)&xwL[nbuf][r][c4 * 4] =
            *(const float4*)(xwc + (size_t)r * Tn * G4 + (size_t)(t + 1) * G4 + c4 * 4);
      }
    }
#endif

    // ---- z += h @ U  via bf16 WMMA, A fragments from LDS ----
    #pragma unroll
    for (int kk = 0; kk < 4; ++kk) {
      const int k0 = kk * 32 + hi * 8;
      Frag16 a;
      a.q[0] = *(const uint4*)(&hB[nl][k0]);
      a.q[1] = *(const uint4*)(&hB[nl][k0 + 16]);
      #pragma unroll
      for (int j = 0; j < 4; ++j) {
        const int n = (wave * 4 + j) * 16 + nl;
        const unsigned short* bp = Ut + (size_t)n * Un + k0;
        Frag16 bf;
        bf.q[0] = *(const uint4*)(bp);
        bf.q[1] = *(const uint4*)(bp + 16);
        acc[j] = __builtin_amdgcn_wmma_f32_16x16x32_bf16(
            false, a.v, false, bf.v, (short)0, acc[j], false, false);
      }
    }

    // ---- spill z tiles to LDS ----
    #pragma unroll
    for (int j = 0; j < 4; ++j) {
      const int n = (wave * 4 + j) * 16 + nl;
      #pragma unroll
      for (int r = 0; r < 8; ++r) zS[r + hi * 8][n] = acc[j][r];
    }
    __syncthreads();

    // ---- gates + masked state update (2048 elems over 256 threads) ----
    for (int e = tid; e < 16 * Un; e += 256) {
      const int bl = e >> 7;
      const int u  = e & 127;
      const float zi = zS[bl][u];
      const float zf = zS[bl][Un + u];
      const float zg = zS[bl][2 * Un + u];
      const float zo = zS[bl][3 * Un + u];
      const float ig = sigm(zi);
      const float fg = sigm(zf);
      const float gg = tanh_fast(zg);
      const float og = sigm(zo);
      const float cold = cS[bl][u];
      const float hold = hS[bl][u];
      const float cn = fg * cold + ig * gg;
      const float hn = og * tanh_fast(cn);
      const size_t row = (size_t)(chunk * 16 + bl) * Tn + t;
      const float m = maskf[row];
      const float c2 = (m > 0.5f) ? cn : cold;
      const float h2 = (m > 0.5f) ? hn : hold;
      cS[bl][u] = c2;
      hS[bl][u] = h2;
      hB[bl][u] = f2bf(h2);
      h_hist[row * Un + u] = h2;
    }

#if HAVE_TDM
    // Ensure next tile's DMA has landed before anyone reads it next step.
    if (wave == 0 && (t + 1) < Tn) __builtin_amdgcn_s_wait_tensorcnt(0);
#endif
    __syncthreads();
  }
}

// ---------------------------------------------------------------------------
// Kernel 5: head  out = relu([h, d] @ Wd + bd). One wave per row, coalesced
// float4/float2 loads, wave32 shuffle reduction.
// ---------------------------------------------------------------------------
__global__ void head_kernel(const float* __restrict__ h_hist,
                            const float* __restrict__ d,
                            const float* __restrict__ Wd,
                            const float* __restrict__ bd,
                            float* __restrict__ out) {
  const int row  = blockIdx.x * 8 + (threadIdx.x >> 5);
  const int lane = threadIdx.x & 31;
  const float4 h4 = *(const float4*)(h_hist + (size_t)row * Un + lane * 4);
  const float4 w4 = *(const float4*)(Wd + lane * 4);
  float s = h4.x * w4.x + h4.y * w4.y + h4.z * w4.z + h4.w * w4.w;
  const float2 d2  = *(const float2*)(d + (size_t)row * FDn + lane * 2);
  const float2 wd2 = *(const float2*)(Wd + Un + lane * 2);
  s += d2.x * wd2.x + d2.y * wd2.y;
  #pragma unroll
  for (int o = 16; o > 0; o >>= 1) s += __shfl_xor(s, o);
  if (lane == 0) out[row] = fmaxf(s + bd[0], 0.0f);
}

// ---------------------------------------------------------------------------
extern "C" void kernel_launch(void* const* d_in, const int* in_sizes, int n_in,
                              void* d_out, int out_size, void* d_ws, size_t ws_size,
                              hipStream_t stream) {
  const float* x  = (const float*)d_in[0];   // [B,T,F]
  const float* d  = (const float*)d_in[1];   // [B,T,FD]
  const float* W  = (const float*)d_in[2];   // [F,4U]
  const float* Uw = (const float*)d_in[3];   // [U,4U]
  const float* b  = (const float*)d_in[4];   // [4U]
  const float* Wd = (const float*)d_in[5];   // [U+FD,1]
  const float* bd = (const float*)d_in[6];   // [1]
  float* out = (float*)d_out;                // [B,T,1]

  char* ws = (char*)d_ws;
  // Workspace layout (256B-aligned offsets)
  unsigned short* xB    = (unsigned short*)(ws);                       // 32 MB
  unsigned short* Wt    = (unsigned short*)(ws + 33554432);            // 128 KB
  unsigned short* Ut    = (unsigned short*)(ws + 33685504);            // 128 KB
  float*          maskf = (float*)(ws + 33816576);                     // 512 KB
  float*          xW    = (float*)(ws + 34340864);                     // 256 MB
  float*          hH    = (float*)(ws + 302776320);                    // 64 MB

  // 1) mask + bf16 convert of x
  prep_x_kernel<<<BT / 8, 256, 0, stream>>>(x, xB, maskf);
  // 2) transpose+convert W, U
  prep_wu_kernel<<<512, 256, 0, stream>>>(W, Uw, Wt, Ut);
  // 3) xW = x @ W + b  (WMMA bf16, fp32 accum)
  xw_gemm_kernel<<<BT / 16, 256, 0, stream>>>(xB, Wt, b, xW);
  // 4) sequential LSTM recurrence, batch-parallel across 4 persistent blocks,
  //    with TDM double-buffered pre-activation tiles in LDS
  lstm_recur_kernel<<<4, 256, 0, stream>>>(xW, Ut, maskf, hH);
  // 5) dense head + relu
  head_kernel<<<BT / 8, 256, 0, stream>>>(hH, d, Wd, bd, out);
}
